// EquivariantMessagePassing_27891517620925
// MI455X (gfx1250) — matrix operand order repaired and checked by reading, not compiled, not run on hardware
//
#include <hip/hip_runtime.h>
#include <hip/hip_bf16.h>

typedef __bf16 bf16;
typedef __attribute__((ext_vector_type(16))) __bf16 v16bf;
typedef __attribute__((ext_vector_type(8)))  __bf16 v8bf;
typedef __attribute__((ext_vector_type(8)))  float   v8f;

#define IN_STRIDE 296   // 288 padded to keep 16B alignment + spread LDS banks
#define H_STRIDE  136   // 128 padded
#define NI_STRIDE 264   // 256 padded
#define A_STRIDE  72    // 64 padded
#define HID       128
#define DIN       288   // 2*H + NB
#define NBASIS    32

__device__ __forceinline__ int laneid() { return threadIdx.x & 31; }

__device__ __forceinline__ float silu_f(float v) { return v / (1.f + __expf(-v)); }

// A fragment (16x32 bf16, row-major source in LDS) per ISA 16-bit A layout:
// lane l (0-15): row M=l, K = kb..kb+7 and kb+16..kb+23, kb = 8*(l>>4)
__device__ __forceinline__ v16bf a_frag(const bf16* tile, int stride, int kt) {
  int l = laneid();
  const bf16* p = tile + (l & 15) * stride + kt * 32 + ((l >> 4) << 3);
  v8bf lo = *(const v8bf*)p;
  v8bf hi = *(const v8bf*)(p + 16);
  v16bf r;
#pragma unroll
  for (int j = 0; j < 8; ++j) { r[j] = lo[j]; r[j + 8] = hi[j]; }
  return r;
}

// B fragment: weights pre-swizzled so each lane reads 32 contiguous bytes.
__device__ __forceinline__ v16bf b_frag(const bf16* w, int t, int kt, int KT) {
  const bf16* p = w + (size_t)((t * KT + kt) * 512 + laneid() * 16);
  return *(const v16bf*)p;
}

__device__ __forceinline__ v8f wmma_bf16(v16bf a, v16bf b, v8f c) {
  return __builtin_amdgcn_wmma_f32_16x16x32_bf16(false, a, false, b, (short)0, c,
                                                 false, false);
}

// Dense layer over a 32-row tile (two 16-row M blocks sharing each B fragment):
// out[32][NT*16] = act(in[32][KT*32] @ W + b).  Two independent accumulator
// chains per (t) hide the WMMA->WMMA RAW latency and halve B traffic.
__device__ __forceinline__ void mlp_layer2(const bf16* in, int inStride, int KT,
                                           bf16* out, int outStride, int NT,
                                           const bf16* W, const float* bias, bool act) {
  int l = laneid();
  int mh = (l >> 4) << 3;
  const bf16* in1 = in + 16 * inStride;
  for (int t = 0; t < NT; ++t) {
    v8f acc0 = {}, acc1 = {};
    for (int kt = 0; kt < KT; ++kt) {
      v16bf b = b_frag(W, t, kt, KT);
      acc0 = wmma_bf16(a_frag(in, inStride, kt), b, acc0);
      acc1 = wmma_bf16(a_frag(in1, inStride, kt), b, acc1);
    }
    int n = t * 16 + (l & 15);
    float bn = bias[n];
#pragma unroll
    for (int i = 0; i < 8; ++i) {
      float v0 = acc0[i] + bn;
      float v1 = acc1[i] + bn;
      if (act) { v0 = silu_f(v0); v1 = silu_f(v1); }
      out[(mh + i) * outStride + n] = (bf16)v0;
      out[(16 + mh + i) * outStride + n] = (bf16)v1;
    }
  }
  __syncthreads();
}

// Gather [x[row] | x[col] | rbf] for 32 edges into LDS as bf16.
__device__ __forceinline__ void build_in_tile32(int eb, const int* erow, const int* ecol,
                                                const bf16* xbf, const float* rbf,
                                                bf16* tile, int* row_s, int* col_s) {
  int l = threadIdx.x;
  row_s[l] = erow[eb + l];
  col_s[l] = ecol[eb + l];
  __syncthreads();
  for (int c = l; c < 32 * 36; c += 32) {
    int e = c / 36, f = c % 36;
    bf16* dst = tile + e * IN_STRIDE + f * 8;
    if (f < 16) {
      *(v8bf*)dst = *(const v8bf*)(xbf + (size_t)row_s[e] * HID + f * 8);
    } else if (f < 32) {
      *(v8bf*)dst = *(const v8bf*)(xbf + (size_t)col_s[e] * HID + (f - 16) * 8);
    } else {
      const float* src = rbf + (size_t)(eb + e) * NBASIS + (f - 32) * 8;
#pragma unroll
      for (int j = 0; j < 8; ++j) dst[j] = (bf16)src[j];
    }
  }
  __syncthreads();
}

// ---------------------------------------------------------------- prep kernels

__global__ __launch_bounds__(256) void zero_f32_kernel(float* p, size_t n) {
  size_t i = (size_t)blockIdx.x * blockDim.x + threadIdx.x;
  if (i < n) p[i] = 0.f;
}

__global__ __launch_bounds__(256) void cvt_bf16_kernel(const float* src, bf16* dst, size_t n) {
  size_t i = (size_t)blockIdx.x * blockDim.x + threadIdx.x;
  if (i < n) dst[i] = (bf16)src[i];
}

// Swizzle W[K][Nout] (fp32 row-major) into per-lane B fragments:
// dst[((t*KT+kt)*32 + lane)*16 + j] = W[kt*32 + 8*(lane>>4) + (j&7) + 16*(j>>3)][t*16 + (lane&15)]
__global__ __launch_bounds__(256) void swizzle_w_kernel(const float* W, bf16* dst,
                                                        int K, int Nout) {
  int idx = blockIdx.x * blockDim.x + threadIdx.x;
  int total = K * Nout;
  if (idx >= total) return;
  int j = idx & 15;
  int lane = (idx >> 4) & 31;
  int ft = idx >> 9;
  int KT = K >> 5;
  int kt = ft % KT;
  int t = ft / KT;
  int n = t * 16 + (lane & 15);
  int k = kt * 32 + ((lane >> 4) << 3) + (j & 7) + ((j >> 3) << 4);
  dst[idx] = (bf16)W[(size_t)k * Nout + n];
}

// ---------------------------------------------------------------- edge pass 1

__global__ __launch_bounds__(32) void edge_attn_kernel(
    const int* erow, const int* ecol, const bf16* xbf, const float* rbf,
    const bf16* wa0, const float* ba0, const float* wa1, const float* ba1,
    float* logits, float* maxv) {
  __shared__ __align__(16) bf16 tile[32 * IN_STRIDE];
  __shared__ __align__(16) bf16 a1[32 * A_STRIDE];
  __shared__ int row_s[32], col_s[32];
  int eb = blockIdx.x * 32;
  build_in_tile32(eb, erow, ecol, xbf, rbf, tile, row_s, col_s);
  mlp_layer2(tile, IN_STRIDE, DIN / 32, a1, A_STRIDE, 4, wa0, ba0, true);
  int l = threadIdx.x;
  float s = ba1[0];
  const bf16* r = a1 + l * A_STRIDE;
  for (int k = 0; k < 64; ++k) s += (float)r[k] * wa1[k];
  logits[eb + l] = s;
  // max starts at 0 (include_self over zeros) -> only positives matter,
  // and positive-float bits compare monotonically as ints.
  if (s > 0.f) atomicMax((int*)(maxv + col_s[l]), __float_as_int(s));
}

// ---------------------------------------------------------------- edge pass 2

__global__ __launch_bounds__(256) void edge_sum_kernel(const int* ecol, const float* logits,
                                                       const float* maxv, float* sumv, int E) {
  int e = blockIdx.x * blockDim.x + threadIdx.x;
  if (e < E) {
    int c = ecol[e];
    atomicAdd(sumv + c, __expf(logits[e] - maxv[c]));
  }
}

// ---------------------------------------------------------------- edge pass 3

__global__ __launch_bounds__(32) void edge_msg_kernel(
    const int* erow, const int* ecol, const bf16* xbf, const float* rbf, const float* pos,
    const bf16* wm0, const float* bm0, const bf16* wm1, const float* bm1,
    const bf16* wm2, const float* bm2,
    const bf16* wc0, const float* bc0, const bf16* wc1, const float* bc1,
    const float* wc2, const float* bc2,
    const float* logits, const float* maxv, const float* sumv,
    float* msg_aggr, float* coord_aggr) {
  __shared__ __align__(16) bf16 tile[32 * IN_STRIDE];
  __shared__ __align__(16) bf16 h1[32 * H_STRIDE];
  __shared__ __align__(16) bf16 h2[32 * H_STRIDE];
  __shared__ int row_s[32], col_s[32];
  __shared__ float attn_s[32];
  int eb = blockIdx.x * 32;
  build_in_tile32(eb, erow, ecol, xbf, rbf, tile, row_s, col_s);
  int l = threadIdx.x;
  {
    int c = col_s[l];
    float ex = __expf(logits[eb + l] - maxv[c]);
    attn_s[l] = ex / (sumv[c] + 1e-8f);
  }
  __syncthreads();

  // ---- scalar message MLP: 288 -> 128 -> 128 -> 128, scaled scatter-add ----
  mlp_layer2(tile, IN_STRIDE, DIN / 32, h1, H_STRIDE, 8, wm0, bm0, true);
  mlp_layer2(h1, H_STRIDE, 4, h2, H_STRIDE, 8, wm1, bm1, true);
  int mh = (l >> 4) << 3;
  for (int t = 0; t < 8; ++t) {
    v8f acc0 = {}, acc1 = {};
    for (int kt = 0; kt < 4; ++kt) {
      v16bf b = b_frag(wm2, t, kt, 4);
      acc0 = wmma_bf16(a_frag(h2, H_STRIDE, kt), b, acc0);
      acc1 = wmma_bf16(a_frag(h2 + 16 * H_STRIDE, H_STRIDE, kt), b, acc1);
    }
    int n = t * 16 + (l & 15);
    float bn = bm2[n];
#pragma unroll
    for (int i = 0; i < 8; ++i) {
      int m0 = mh + i;
      int m1 = 16 + m0;
      atomicAdd(msg_aggr + (size_t)col_s[m0] * HID + n, (acc0[i] + bn) * attn_s[m0]);
      atomicAdd(msg_aggr + (size_t)col_s[m1] * HID + n, (acc1[i] + bn) * attn_s[m1]);
    }
  }
  __syncthreads();

  // ---- coord MLP: 288 -> 128 -> 128 -> 1, scatter-add w * edge_dir ----
  mlp_layer2(tile, IN_STRIDE, DIN / 32, h1, H_STRIDE, 8, wc0, bc0, true);
  mlp_layer2(h1, H_STRIDE, 4, h2, H_STRIDE, 8, wc1, bc1, true);
  {
    float w = bc2[0];
    const bf16* r = h2 + l * H_STRIDE;
    for (int k = 0; k < HID; ++k) w += (float)r[k] * wc2[k];
    int ri = row_s[l], ci = col_s[l];
    float dx = pos[ci * 3 + 0] - pos[ri * 3 + 0];
    float dy = pos[ci * 3 + 1] - pos[ri * 3 + 1];
    float dz = pos[ci * 3 + 2] - pos[ri * 3 + 2];
    float inv = w / (sqrtf(dx * dx + dy * dy + dz * dz) + 1e-8f);
    atomicAdd(coord_aggr + ci * 3 + 0, inv * dx);
    atomicAdd(coord_aggr + ci * 3 + 1, inv * dy);
    atomicAdd(coord_aggr + ci * 3 + 2, inv * dz);
  }
}

// ---------------------------------------------------------------- node update

__global__ __launch_bounds__(32) void node_kernel(
    const bf16* xbf, const float* x, const float* msg_aggr,
    const bf16* wn0, const float* bn0, const bf16* wn1, const float* bn1,
    const float* gamma, const float* beta, float* xout, int N) {
  __shared__ __align__(16) bf16 tile[32 * NI_STRIDE];
  __shared__ __align__(16) bf16 u1[32 * H_STRIDE];
  __shared__ __align__(16) float ybuf[32 * 132];
  int nb = blockIdx.x * 32;
  int rows = N - nb;
  if (rows > 32) rows = 32;
  int l = threadIdx.x;
  for (int c = l; c < 32 * 32; c += 32) {
    int m = c / 32, f = c % 32;
    if (m >= rows) continue;
    bf16* dst = tile + m * NI_STRIDE + f * 8;
    if (f < 16) {
      *(v8bf*)dst = *(const v8bf*)(xbf + (size_t)(nb + m) * HID + f * 8);
    } else {
      const float* src = msg_aggr + (size_t)(nb + m) * HID + (f - 16) * 8;
#pragma unroll
      for (int j = 0; j < 8; ++j) dst[j] = (bf16)src[j];
    }
  }
  __syncthreads();
  mlp_layer2(tile, NI_STRIDE, 8, u1, H_STRIDE, 8, wn0, bn0, true);
  int mh = (l >> 4) << 3;
  for (int t = 0; t < 8; ++t) {
    v8f acc0 = {}, acc1 = {};
    for (int kt = 0; kt < 4; ++kt) {
      v16bf b = b_frag(wn1, t, kt, 4);
      acc0 = wmma_bf16(a_frag(u1, H_STRIDE, kt), b, acc0);
      acc1 = wmma_bf16(a_frag(u1 + 16 * H_STRIDE, H_STRIDE, kt), b, acc1);
    }
    int n = t * 16 + (l & 15);
    float bn = bn1[n];
#pragma unroll
    for (int i = 0; i < 8; ++i) {
      int m0 = mh + i;
      int m1 = 16 + m0;
      if (m0 < rows)
        ybuf[m0 * 132 + n] = acc0[i] + bn + x[(size_t)(nb + m0) * HID + n];
      if (m1 < rows)
        ybuf[m1 * 132 + n] = acc1[i] + bn + x[(size_t)(nb + m1) * HID + n];
    }
  }
  __syncthreads();
  if (l < rows) {
    const float* yr = ybuf + l * 132;
    float mu = 0.f;
    for (int k = 0; k < HID; ++k) mu += yr[k];
    mu *= (1.f / HID);
    float var = 0.f;
    for (int k = 0; k < HID; ++k) { float d = yr[k] - mu; var += d * d; }
    var *= (1.f / HID);
    float inv = rsqrtf(var + 1e-5f);
    float* o = xout + (size_t)(nb + l) * HID;
    for (int k = 0; k < HID; ++k) o[k] = (yr[k] - mu) * inv * gamma[k] + beta[k];
  }
}

__global__ __launch_bounds__(256) void pos_kernel(const float* pos, const float* coord_aggr,
                                                  float* pout, int n) {
  int i = blockIdx.x * blockDim.x + threadIdx.x;
  if (i < n) pout[i] = pos[i] + coord_aggr[i];
}

// ---------------------------------------------------------------- host launch

extern "C" void kernel_launch(void* const* d_in, const int* in_sizes, int n_in,
                              void* d_out, int out_size, void* d_ws, size_t ws_size,
                              hipStream_t stream) {
  (void)n_in; (void)out_size; (void)ws_size;
  const int H = HID;
  const int N = in_sizes[0] / H;
  const int E = in_sizes[2] / 2;

  const float* x   = (const float*)d_in[0];
  const float* pos = (const float*)d_in[1];
  const int*   erow = (const int*)d_in[2];   // JAX x64 disabled -> int32
  const int*   ecol = erow + E;
  const float* rbf = (const float*)d_in[3];

  auto F = [&](int i) { return (const float*)d_in[i]; };
  const float *msgW0, *msgb0, *msgW1, *msgb1, *msgW2, *msgb2;
  const float *attW0, *attb0, *attW1, *attb1;
  const float *cW0, *cb0, *cW1, *cb1, *cW2, *cb2;
  const float *nW0, *nb0, *nW1, *nb1, *gamma, *beta;
  if (in_sizes[4] == DIN * 64) {  // alphabetical (jax tree) order: attn,coord,ln,msg,node
    attW0 = F(4);  attb0 = F(5);  attW1 = F(6);  attb1 = F(7);
    cW0 = F(8);  cb0 = F(9);  cW1 = F(10); cb1 = F(11); cW2 = F(12); cb2 = F(13);
    gamma = F(14); beta = F(15);
    msgW0 = F(16); msgb0 = F(17); msgW1 = F(18); msgb1 = F(19); msgW2 = F(20); msgb2 = F(21);
    nW0 = F(22); nb0 = F(23); nW1 = F(24); nb1 = F(25);
  } else {  // dict insertion order: msg,attn,coord,node,ln
    msgW0 = F(4);  msgb0 = F(5);  msgW1 = F(6);  msgb1 = F(7);  msgW2 = F(8); msgb2 = F(9);
    attW0 = F(10); attb0 = F(11); attW1 = F(12); attb1 = F(13);
    cW0 = F(14); cb0 = F(15); cW1 = F(16); cb1 = F(17); cW2 = F(18); cb2 = F(19);
    nW0 = F(20); nb0 = F(21); nW1 = F(22); nb1 = F(23);
    gamma = F(24); beta = F(25);
  }

  // ---- workspace layout ----
  char* ws = (char*)d_ws;
  size_t off = 0;
  auto take = [&](size_t bytes) -> void* {
    void* p = ws + off;
    off = (off + bytes + 255) & ~(size_t)255;
    return p;
  };
  bf16* xbf = (bf16*)take((size_t)N * H * 2);
  // contiguous zero region: msg_aggr | coord_aggr | maxv | sumv
  size_t zcount = (size_t)N * H + (size_t)N * 3 + N + N;
  float* zbase = (float*)take(zcount * 4);
  float* msg_aggr   = zbase;
  float* coord_aggr = zbase + (size_t)N * H;
  float* maxv       = coord_aggr + (size_t)N * 3;
  float* sumv       = maxv + N;
  float* logits = (float*)take((size_t)E * 4);
  bf16* wm0 = (bf16*)take((size_t)DIN * 128 * 2);
  bf16* wm1 = (bf16*)take((size_t)128 * 128 * 2);
  bf16* wm2 = (bf16*)take((size_t)128 * 128 * 2);
  bf16* wa0 = (bf16*)take((size_t)DIN * 64 * 2);
  bf16* wc0 = (bf16*)take((size_t)DIN * 128 * 2);
  bf16* wc1 = (bf16*)take((size_t)128 * 128 * 2);
  bf16* wn0 = (bf16*)take((size_t)256 * 128 * 2);
  bf16* wn1 = (bf16*)take((size_t)128 * 128 * 2);

  float* x_out = (float*)d_out;
  float* p_out = x_out + (size_t)N * H;

  // ---- prep ----
  {
    size_t g = (zcount + 255) / 256;
    zero_f32_kernel<<<(unsigned)g, 256, 0, stream>>>(zbase, zcount);
    size_t nx = (size_t)N * H;
    cvt_bf16_kernel<<<(unsigned)((nx + 255) / 256), 256, 0, stream>>>(x, xbf, nx);
    auto sw = [&](const float* src, bf16* dst, int K, int Nout) {
      int tot = K * Nout;
      swizzle_w_kernel<<<(tot + 255) / 256, 256, 0, stream>>>(src, dst, K, Nout);
    };
    sw(msgW0, wm0, DIN, 128);
    sw(msgW1, wm1, 128, 128);
    sw(msgW2, wm2, 128, 128);
    sw(attW0, wa0, DIN, 64);
    sw(cW0, wc0, DIN, 128);
    sw(cW1, wc1, 128, 128);
    sw(nW0, wn0, 256, 128);
    sw(nW1, wn1, 128, 128);
  }

  int etiles = E / 32;             // 400000 -> 12500
  int ntiles = (N + 31) / 32;      // 50000  -> 1563 (tail-guarded)

  edge_attn_kernel<<<etiles, 32, 0, stream>>>(erow, ecol, xbf, rbf, wa0, attb0,
                                              attW1, attb1, logits, maxv);
  edge_sum_kernel<<<(E + 255) / 256, 256, 0, stream>>>(ecol, logits, maxv, sumv, E);
  edge_msg_kernel<<<etiles, 32, 0, stream>>>(erow, ecol, xbf, rbf, pos,
                                             wm0, msgb0, wm1, msgb1, wm2, msgb2,
                                             wc0, cb0, wc1, cb1, cW2, cb2,
                                             logits, maxv, sumv, msg_aggr, coord_aggr);
  node_kernel<<<ntiles, 32, 0, stream>>>(xbf, x, msg_aggr, wn0, nb0, wn1, nb1,
                                         gamma, beta, x_out, N);
  pos_kernel<<<(N * 3 + 255) / 256, 256, 0, stream>>>(pos, coord_aggr, p_out, N * 3);
}
